// Net_GAT_Conv_SoA_1_58746562675274
// MI455X (gfx1250) — matrix operand order repaired and checked by reading, not compiled, not run on hardware
//
#include <hip/hip_runtime.h>
#include <hip/hip_bf16.h>

// ---------------------------------------------------------------------------
// Types for CDNA5 WMMA (wave32): V_WMMA_F32_16X16X32_F16
// ---------------------------------------------------------------------------
typedef __attribute__((ext_vector_type(16))) _Float16 v16h;
typedef __attribute__((ext_vector_type(8)))  _Float16 v8h;
typedef __attribute__((ext_vector_type(8)))  float    v8f;

union AFragU { v16h v; v8h h[2]; };

// A fragment (16x32 f16, MxK). lane = hlf*16 + m.
// elements 0..7 : K = kbase + hlf*16 + e       (contiguous 16B)
// elements 8..15: K = kbase + hlf*16 + 8 + e   (contiguous 16B)
__device__ __forceinline__ v16h load_a_frag(const _Float16* __restrict__ Arow,
                                            int kbase, int hlf) {
  AFragU u;
  u.h[0] = *(const v8h*)(Arow + kbase + hlf * 16);
  u.h[1] = *(const v8h*)(Arow + kbase + hlf * 16 + 8);
  return u.v;
}

// B fragment (32x16 f16, KxN) with B pre-transposed to [N,K] row-major.
// lane = hlf*16 + n ; elements 0..15 : K = kbase + hlf*16 + e (contiguous 32B)
__device__ __forceinline__ v16h load_b_frag(const _Float16* __restrict__ Bcol,
                                            int kbase, int hlf) {
  return *(const v16h*)(Bcol + kbase + hlf * 16);
}

// ---------------------------------------------------------------------------
// Generic WMMA GEMM:  C[M,N] = A[M,K](f16,row) * Bt[N,K](f16,row = B^T)
// block = 256 threads = 8 waves; wave w -> rows [by*128 + w*16, +16), cols [bx*64, +64)
// EPI 0: store f32. EPI 1: ReLU then BatchNorm (eps=0.1), store f32 (if Cf) and f16 (if Ch).
// ---------------------------------------------------------------------------
template <int EPI>
__global__ void gemm_wmma(const _Float16* __restrict__ A,
                          const _Float16* __restrict__ Bt,
                          int M, int N, int K,
                          float* __restrict__ Cf, _Float16* __restrict__ Ch,
                          const float* __restrict__ bn_g, const float* __restrict__ bn_b,
                          const float* __restrict__ bn_m, const float* __restrict__ bn_v) {
  const int lane = threadIdx.x & 31;
  const int wave = threadIdx.x >> 5;
  const int hlf  = lane >> 4;
  const int l16  = lane & 15;
  const int row  = blockIdx.y * 128 + wave * 16 + l16;
  const int colBase = blockIdx.x * 64;

  const _Float16* Arow = A + (size_t)row * K;
  const _Float16* Bp0 = Bt + (size_t)(colBase + 0 * 16 + l16) * K;
  const _Float16* Bp1 = Bt + (size_t)(colBase + 1 * 16 + l16) * K;
  const _Float16* Bp2 = Bt + (size_t)(colBase + 2 * 16 + l16) * K;
  const _Float16* Bp3 = Bt + (size_t)(colBase + 3 * 16 + l16) * K;

  v8f acc0 = {}, acc1 = {}, acc2 = {}, acc3 = {};
  for (int kb = 0; kb < K; kb += 32) {
    v16h a  = load_a_frag(Arow, kb, hlf);
    v16h b0 = load_b_frag(Bp0, kb, hlf);
    v16h b1 = load_b_frag(Bp1, kb, hlf);
    v16h b2 = load_b_frag(Bp2, kb, hlf);
    v16h b3 = load_b_frag(Bp3, kb, hlf);
    acc0 = __builtin_amdgcn_wmma_f32_16x16x32_f16(false, a, false, b0, (short)0, acc0, false, false);
    acc1 = __builtin_amdgcn_wmma_f32_16x16x32_f16(false, a, false, b1, (short)0, acc1, false, false);
    acc2 = __builtin_amdgcn_wmma_f32_16x16x32_f16(false, a, false, b2, (short)0, acc2, false, false);
    acc3 = __builtin_amdgcn_wmma_f32_16x16x32_f16(false, a, false, b3, (short)0, acc3, false, false);
  }

  const int outRow = blockIdx.y * 128 + wave * 16 + hlf * 8; // D: M = e + 8*hlf
  v8f accs[4] = {acc0, acc1, acc2, acc3};
#pragma unroll
  for (int t = 0; t < 4; ++t) {
    int col = colBase + t * 16 + l16;
    float scale = 1.f, shift = 0.f;
    if (EPI == 1) {
      float inv = rsqrtf(bn_v[col] + 0.1f);
      scale = bn_g[col] * inv;
      shift = bn_b[col] - bn_m[col] * scale;
    }
#pragma unroll
    for (int e = 0; e < 8; ++e) {
      float val = accs[t][e];
      if (EPI == 1) { val = val > 0.f ? val : 0.f; val = val * scale + shift; }
      size_t off = (size_t)(outRow + e) * N + col;
      if (Cf) Cf[off] = val;
      if (EPI == 1 && Ch) Ch[off] = (_Float16)val;
    }
  }
}

// ---------------------------------------------------------------------------
// LSTM cell GEMM: Z[128,N] = A1[128,K1]@B1t^T + A2[128,K2]@B2t^T + bi + bh
// (Wih/Whh are stored [4H, In] row-major which IS the transposed-B layout.)
// ---------------------------------------------------------------------------
__global__ void lstm_gemm(const _Float16* __restrict__ A1, const _Float16* __restrict__ B1t, int K1,
                          const _Float16* __restrict__ A2, const _Float16* __restrict__ B2t, int K2,
                          int N, const float* __restrict__ bi, const float* __restrict__ bh,
                          float* __restrict__ Z) {
  const int lane = threadIdx.x & 31;
  const int wave = threadIdx.x >> 5;
  const int hlf  = lane >> 4;
  const int l16  = lane & 15;
  const int row  = wave * 16 + l16;          // M = 128 exactly, grid.y = 1
  const int colBase = blockIdx.x * 64;

  const _Float16* A1row = A1 + (size_t)row * K1;
  const _Float16* A2row = A2 + (size_t)row * K2;

  v8f acc[4] = {};
#pragma unroll
  for (int t = 0; t < 4; ++t) {
    const _Float16* Bp = B1t + (size_t)(colBase + t * 16 + l16) * K1;
    v8f a_ = acc[t];
    for (int kb = 0; kb < K1; kb += 32) {
      v16h a = load_a_frag(A1row, kb, hlf);
      v16h b = load_b_frag(Bp, kb, hlf);
      a_ = __builtin_amdgcn_wmma_f32_16x16x32_f16(false, a, false, b, (short)0, a_, false, false);
    }
    acc[t] = a_;
  }
#pragma unroll
  for (int t = 0; t < 4; ++t) {
    const _Float16* Bp = B2t + (size_t)(colBase + t * 16 + l16) * K2;
    v8f a_ = acc[t];
    for (int kb = 0; kb < K2; kb += 32) {
      v16h a = load_a_frag(A2row, kb, hlf);
      v16h b = load_b_frag(Bp, kb, hlf);
      a_ = __builtin_amdgcn_wmma_f32_16x16x32_f16(false, a, false, b, (short)0, a_, false, false);
    }
    acc[t] = a_;
  }
  const int outRow = wave * 16 + hlf * 8;
#pragma unroll
  for (int t = 0; t < 4; ++t) {
    int col = colBase + t * 16 + l16;
    float bsum = bi[col] + bh[col];
#pragma unroll
    for (int e = 0; e < 8; ++e)
      Z[(size_t)(outRow + e) * N + col] = acc[t][e] + bsum;
  }
}

// ---------------------------------------------------------------------------
// LSTM pointwise: gates i,f,g,o -> update c,h ; emit h as f16 too.
// ---------------------------------------------------------------------------
__global__ void lstm_pointwise(const float* __restrict__ Z, float* __restrict__ h,
                               float* __restrict__ c, _Float16* __restrict__ hh) {
  int idx = blockIdx.x * 256 + threadIdx.x;     // 128*1024
  int b = idx >> 10, j = idx & 1023;
  const float* zr = Z + (size_t)b * 4096;
  float zi = zr[j], zf = zr[1024 + j], zg = zr[2048 + j], zo = zr[3072 + j];
  float si = 1.f / (1.f + __expf(-zi));
  float sf = 1.f / (1.f + __expf(-zf));
  float so = 1.f / (1.f + __expf(-zo));
  float cn = sf * c[idx] + si * tanhf(zg);
  c[idx] = cn;
  float hn = so * tanhf(cn);
  h[idx] = hn;
  hh[idx] = (_Float16)hn;
}

// ---------------------------------------------------------------------------
// GAT per-node attention scores: es/ed[n, gat*8+head] = h[n,head,:]·a
// ---------------------------------------------------------------------------
__global__ void gat_scores(const float* __restrict__ hbuf,
                           const float* __restrict__ as1, const float* __restrict__ ad1,
                           const float* __restrict__ as2, const float* __restrict__ ad2,
                           float* __restrict__ es, float* __restrict__ ed) {
  int idx = blockIdx.x * 256 + threadIdx.x;     // 8192*16
  int n = idx >> 4, gh = idx & 15;
  int gat = gh >> 3, head = gh & 7;
  const float* a_s = (gat ? as2 : as1) + head * 64;
  const float* a_d = (gat ? ad2 : ad1) + head * 64;
  const float* hrow = hbuf + (size_t)n * 1024 + gat * 512 + head * 64;
  float ss = 0.f, sd = 0.f;
#pragma unroll 8
  for (int d = 0; d < 64; ++d) { float hv = hrow[d]; ss += hv * a_s[d]; sd += hv * a_d[d]; }
  es[idx] = ss; ed[idx] = sd;
}

// ---------------------------------------------------------------------------
// GAT softmax + aggregation per (graph, gat). Loops over 8 heads.
// e[i,j] = leaky_relu(ed[i]+es[j], 0.2) masked by |pos_i-pos_j|^2<=16 ; softmax_j ;
// out[i,d] = relu(sum_j alpha[i,j]*h[j,d] + bias)
// ---------------------------------------------------------------------------
__global__ void gat_aggregate(const float* __restrict__ hbuf,
                              const float* __restrict__ es, const float* __restrict__ ed,
                              const float* __restrict__ pos,
                              const float* __restrict__ bias1, const float* __restrict__ bias2,
                              float* __restrict__ x2) {
  int b = blockIdx.x, gat = blockIdx.y, tid = threadIdx.x;
  __shared__ float px[64], py[64], esS[64], edS[64];
  __shared__ float hS[64][64];
  __shared__ float alpha[64][64];
  __shared__ float red[64][4];
  const float* bias = gat ? bias2 : bias1;
  const int nbase = b * 64;
  if (tid < 64) { px[tid] = pos[(nbase + tid) * 2]; py[tid] = pos[(nbase + tid) * 2 + 1]; }
  __syncthreads();
  for (int head = 0; head < 8; ++head) {
    for (int idx = tid; idx < 4096; idx += 256) {
      int j = idx >> 6, d = idx & 63;
      hS[j][d] = hbuf[(size_t)(nbase + j) * 1024 + gat * 512 + head * 64 + d];
    }
    if (tid < 64) {
      esS[tid] = es[(nbase + tid) * 16 + gat * 8 + head];
      edS[tid] = ed[(nbase + tid) * 16 + gat * 8 + head];
    }
    __syncthreads();
    int i = tid >> 2, part = tid & 3;
    float xi = px[i], yi = py[i], edi = edS[i];
    float lmax = -1e30f;
#pragma unroll
    for (int q = 0; q < 16; ++q) {
      int j = part * 16 + q;
      float dx = xi - px[j], dy = yi - py[j];
      bool adj = (dx * dx + dy * dy) <= 16.0f;
      float v = edi + esS[j];
      v = v > 0.f ? v : 0.2f * v;
      v = adj ? v : -1e30f;
      alpha[i][j] = v;
      lmax = fmaxf(lmax, v);
    }
    red[i][part] = lmax;
    __syncthreads();
    float rmax = fmaxf(fmaxf(red[i][0], red[i][1]), fmaxf(red[i][2], red[i][3]));
    __syncthreads();
    float lsum = 0.f;
#pragma unroll
    for (int q = 0; q < 16; ++q) {
      int j = part * 16 + q;
      float e2 = __expf(alpha[i][j] - rmax);
      alpha[i][j] = e2; lsum += e2;
    }
    red[i][part] = lsum;
    __syncthreads();
    float rinv = 1.f / (red[i][0] + red[i][1] + red[i][2] + red[i][3]);
#pragma unroll
    for (int q = 0; q < 16; ++q) alpha[i][part * 16 + q] *= rinv;
    __syncthreads();
    for (int idx = tid; idx < 4096; idx += 256) {
      int ii = idx >> 6, d = idx & 63;
      float s = 0.f;
#pragma unroll 8
      for (int j = 0; j < 64; ++j) s += alpha[ii][j] * hS[j][d];
      float o = s + bias[head * 64 + d];
      x2[(size_t)(nbase + ii) * 1024 + gat * 512 + head * 64 + d] = o > 0.f ? o : 0.f;
    }
    __syncthreads();
  }
}

// ---------------------------------------------------------------------------
// Set2Set attention: e = x2·h1, softmax over 64 nodes, r = sum a*x2 ; q* = [h1|r]
// One block per graph.
// ---------------------------------------------------------------------------
__global__ void s2s_attn(const float* __restrict__ x2, const float* __restrict__ h1,
                         float* __restrict__ qs, _Float16* __restrict__ qsh) {
  int b = blockIdx.x, tid = threadIdx.x;
  __shared__ float hs[1024];
  __shared__ float red[256];
  __shared__ float aw[64];
  for (int d = tid; d < 1024; d += 256) hs[d] = h1[(size_t)b * 1024 + d];
  __syncthreads();
  int n = tid >> 2, part = tid & 3;
  const float* xrow = x2 + (size_t)(b * 64 + n) * 1024;
  float s = 0.f;
  for (int d = part * 256; d < part * 256 + 256; ++d) s += xrow[d] * hs[d];
  red[tid] = s;
  __syncthreads();
  if (part == 0) aw[n] = red[tid] + red[tid + 1] + red[tid + 2] + red[tid + 3];
  __syncthreads();
  if (tid == 0) {
    float mx = -1e30f;
    for (int q = 0; q < 64; ++q) mx = fmaxf(mx, aw[q]);
    float sm = 0.f;
    for (int q = 0; q < 64; ++q) { float e = __expf(aw[q] - mx); aw[q] = e; sm += e; }
    float inv = 1.f / sm;
    for (int q = 0; q < 64; ++q) aw[q] *= inv;
  }
  __syncthreads();
  for (int d = tid; d < 1024; d += 256) {
    float r = 0.f;
    for (int j = 0; j < 64; ++j) r += aw[j] * x2[(size_t)(b * 64 + j) * 1024 + d];
    size_t o0 = (size_t)b * 2048 + d;
    qs[o0] = hs[d];           qs[o0 + 1024] = r;
    qsh[o0] = (_Float16)hs[d]; qsh[o0 + 1024] = (_Float16)r;
  }
}

// ---------------------------------------------------------------------------
// Final head: out[128,10] = y @ W3 + b3
// ---------------------------------------------------------------------------
__global__ void lin3_kernel(const float* __restrict__ y, const float* __restrict__ W3,
                            const float* __restrict__ b3, float* __restrict__ out) {
  int idx = blockIdx.x * 256 + threadIdx.x;
  if (idx >= 1280) return;
  int r = idx / 10, c = idx % 10;
  float s = b3[c];
  for (int k = 0; k < 1024; ++k) s += y[(size_t)r * 1024 + k] * W3[k * 10 + c];
  out[idx] = s;
}

// ---------------------------------------------------------------------------
// Helpers: convert, transpose-convert, zero
// ---------------------------------------------------------------------------
__global__ void cvt_f16(const float* __restrict__ in, _Float16* __restrict__ out, int n) {
  for (int i = blockIdx.x * 256 + threadIdx.x; i < n; i += gridDim.x * 256)
    out[i] = (_Float16)in[i];
}

__global__ void transpose_f32_f16(const float* __restrict__ in, _Float16* __restrict__ out,
                                  int R, int C) {  // out[c*R + r] = in[r*C + c]
  __shared__ float tile[32][33];
  int cb = blockIdx.x * 32, rb = blockIdx.y * 32;
  int tx = threadIdx.x & 31, ty = threadIdx.x >> 5;
  for (int r = ty; r < 32; r += 8) tile[r][tx] = in[(size_t)(rb + r) * C + cb + tx];
  __syncthreads();
  for (int r = ty; r < 32; r += 8) out[(size_t)(cb + r) * R + rb + tx] = (_Float16)tile[tx][r];
}

__global__ void zero_u32(unsigned int* __restrict__ p, int n) {
  for (int i = blockIdx.x * 256 + threadIdx.x; i < n; i += gridDim.x * 256) p[i] = 0u;
}

// ---------------------------------------------------------------------------
// Host launcher
// ---------------------------------------------------------------------------
extern "C" void kernel_launch(void* const* d_in, const int* in_sizes, int n_in,
                              void* d_out, int out_size, void* d_ws, size_t ws_size,
                              hipStream_t stream) {
  const float* x     = (const float*)d_in[0];
  const float* pos   = (const float*)d_in[1];
  const float* W0    = (const float*)d_in[3];
  const float* bn0_g = (const float*)d_in[4];
  const float* bn0_b = (const float*)d_in[5];
  const float* bn0_m = (const float*)d_in[6];
  const float* bn0_v = (const float*)d_in[7];
  const float* Wg1   = (const float*)d_in[8];
  const float* as1   = (const float*)d_in[9];
  const float* ad1   = (const float*)d_in[10];
  const float* b1    = (const float*)d_in[11];
  const float* Wg2   = (const float*)d_in[12];
  const float* as2   = (const float*)d_in[13];
  const float* ad2   = (const float*)d_in[14];
  const float* b2    = (const float*)d_in[15];
  const float* Wih0  = (const float*)d_in[16];
  const float* Whh0  = (const float*)d_in[17];
  const float* bih0  = (const float*)d_in[18];
  const float* bhh0  = (const float*)d_in[19];
  const float* Wih1  = (const float*)d_in[20];
  const float* Whh1  = (const float*)d_in[21];
  const float* bih1  = (const float*)d_in[22];
  const float* bhh1  = (const float*)d_in[23];
  const float* W2    = (const float*)d_in[24];
  const float* bn2_g = (const float*)d_in[25];
  const float* bn2_b = (const float*)d_in[26];
  const float* bn2_m = (const float*)d_in[27];
  const float* bn2_v = (const float*)d_in[28];
  const float* W3    = (const float*)d_in[29];
  const float* b3    = (const float*)d_in[30];
  float* out = (float*)d_out;

  // ---- workspace layout (256B aligned) ----
  char* ws = (char*)d_ws;
  size_t off = 0;
  auto alloc = [&](size_t bytes) { void* p = ws + off; off = (off + bytes + 255) & ~(size_t)255; return p; };

  _Float16* xh    = (_Float16*)alloc((size_t)8192 * 2048 * 2);  // also reused as x2 (f32, same size)
  _Float16* x1h   = (_Float16*)alloc((size_t)8192 * 2048 * 2);
  _Float16* W0t   = (_Float16*)alloc((size_t)2048 * 2048 * 2);
  _Float16* Wgt   = (_Float16*)alloc((size_t)1024 * 2048 * 2);
  float*    hbuf  = (float*)   alloc((size_t)8192 * 1024 * 4);
  float*    esb   = (float*)   alloc((size_t)8192 * 16 * 4);
  float*    edb   = (float*)   alloc((size_t)8192 * 16 * 4);
  _Float16* Wih0h = (_Float16*)alloc((size_t)4096 * 2048 * 2);
  _Float16* Whh0h = (_Float16*)alloc((size_t)4096 * 1024 * 2);
  _Float16* Wih1h = (_Float16*)alloc((size_t)4096 * 1024 * 2);
  _Float16* Whh1h = (_Float16*)alloc((size_t)4096 * 1024 * 2);
  _Float16* W2t   = (_Float16*)alloc((size_t)1024 * 2048 * 2);
  float*    z     = (float*)   alloc((size_t)128 * 4096 * 4);
  char*     state = (char*)    alloc((size_t)4 * 1024 * 1024);
  float*    y     = (float*)   alloc((size_t)128 * 1024 * 4);

  float* x2 = (float*)xh; // alias: xh is dead after GEMM0, x2 written after GAT GEMM

  // state sub-buffers (contiguous 4 MB, zeroed each call)
  float*    qs  = (float*)(state + 0);
  float*    h0  = (float*)(state + 1048576);
  float*    c0  = (float*)(state + 1572864);
  float*    h1  = (float*)(state + 2097152);
  float*    c1  = (float*)(state + 2621440);
  _Float16* qsh = (_Float16*)(state + 3145728);
  _Float16* h0h = (_Float16*)(state + 3670016);
  _Float16* h1h = (_Float16*)(state + 3932160);

  const int T = 256;

  // ---- stage 0: precision staging (f32 -> f16, weights transposed to [N,K]) ----
  cvt_f16<<<4096, T, 0, stream>>>(x, xh, 8192 * 2048);
  transpose_f32_f16<<<dim3(64, 64), T, 0, stream>>>(W0, W0t, 2048, 2048);
  transpose_f32_f16<<<dim3(16, 64), T, 0, stream>>>(Wg1, Wgt, 2048, 512);
  transpose_f32_f16<<<dim3(16, 64), T, 0, stream>>>(Wg2, Wgt + (size_t)512 * 2048, 2048, 512);
  cvt_f16<<<4096, T, 0, stream>>>(Wih0, Wih0h, 4096 * 2048);
  cvt_f16<<<2048, T, 0, stream>>>(Whh0, Whh0h, 4096 * 1024);
  cvt_f16<<<2048, T, 0, stream>>>(Wih1, Wih1h, 4096 * 1024);
  cvt_f16<<<2048, T, 0, stream>>>(Whh1, Whh1h, 4096 * 1024);
  transpose_f32_f16<<<dim3(32, 64), T, 0, stream>>>(W2, W2t, 2048, 1024);
  zero_u32<<<1024, T, 0, stream>>>((unsigned int*)state, 4 * 1024 * 1024 / 4);

  // ---- stage 1: x1 = BN(relu(x @ W0))  [8192,2048], f16 out for next GEMM ----
  gemm_wmma<1><<<dim3(32, 64), T, 0, stream>>>(xh, W0t, 8192, 2048, 2048,
                                               nullptr, x1h, bn0_g, bn0_b, bn0_m, bn0_v);

  // ---- stage 2: h = x1 @ [Wg1|Wg2]  [8192,1024] f32 ----
  gemm_wmma<0><<<dim3(16, 64), T, 0, stream>>>(x1h, Wgt, 8192, 1024, 2048,
                                               hbuf, nullptr, nullptr, nullptr, nullptr, nullptr);

  // ---- stage 3: GAT attention ----
  gat_scores<<<512, T, 0, stream>>>(hbuf, as1, ad1, as2, ad2, esb, edb);
  gat_aggregate<<<dim3(128, 2), T, 0, stream>>>(hbuf, esb, edb, pos, b1, b2, x2);

  // ---- stage 4: Set2Set (2 steps, 2-layer LSTM) ----
  for (int step = 0; step < 2; ++step) {
    lstm_gemm<<<dim3(64, 1), T, 0, stream>>>(qsh, Wih0h, 2048, h0h, Whh0h, 1024,
                                             4096, bih0, bhh0, z);
    lstm_pointwise<<<512, T, 0, stream>>>(z, h0, c0, h0h);
    lstm_gemm<<<dim3(64, 1), T, 0, stream>>>(h0h, Wih1h, 1024, h1h, Whh1h, 1024,
                                             4096, bih1, bhh1, z);
    lstm_pointwise<<<512, T, 0, stream>>>(z, h1, c1, h1h);
    s2s_attn<<<128, T, 0, stream>>>(x2, h1, qs, qsh);
  }

  // ---- stage 5: y = BN(relu(q_star @ W2)) [128,1024], then out = y @ W3 + b3 ----
  gemm_wmma<1><<<dim3(16, 1), T, 0, stream>>>(qsh, W2t, 128, 1024, 2048,
                                              y, nullptr, bn2_g, bn2_b, bn2_m, bn2_v);
  lin3_kernel<<<5, T, 0, stream>>>(y, W3, b3, out);
}
